// Pytorch_DTLN_stateful_13718125544063
// MI455X (gfx1250) — compile-verified, matrix operand order checked by reading, not run on hardware
//
#include <hip/hip_runtime.h>
#include <hip/hip_bf16.h>
#include <math.h>

// ---------------------------------------------------------------------------
// DTLN on gfx1250: everything frame-parallel is a WMMA GEMM (M = 16*1021),
// the GRU recurrences run in a single persistent block using WMMA with
// M=batch=16, Whh fragments held in VGPRs across the 1021-step scan.
// GEMM mainloop: double-buffered GLOBAL_LOAD_ASYNC_TO_LDS_B128 (ASYNCcnt)
// pipelined against v_wmma; falls back to classic ld/st staging if the
// builtin is absent.
// ---------------------------------------------------------------------------

typedef _Float16 h16;
typedef h16   v16h __attribute__((ext_vector_type(16)));
typedef float v8f  __attribute__((ext_vector_type(8)));

#define FRAME_ 1024
#define HOP_   256
#define HID_   128
#define ENC_   256
#define FREQ_  513
#define FP_    528      // FREQ padded to multiple of 16 (N-dim padding)
#define FPK_   544      // FREQ padded to multiple of 32 (K-dim padding)
#define SPECW_ 1056     // [RE(528) | IM(528)]
#define BDIM_  16
#define TLEN_  262144
#define NF_    1021
#define BF_    16336    // BDIM_*NF_
#define PI_    3.14159265358979323846f

#if __has_builtin(__builtin_amdgcn_global_load_async_to_lds_b128)
#define HAS_ASYNC_LDS 1
#else
#define HAS_ASYNC_LDS 0
#endif

#if HAS_ASYNC_LDS
typedef int v4i_t __attribute__((vector_size(16)));
typedef __attribute__((address_space(1))) v4i_t* as1_v4i;
typedef __attribute__((address_space(3))) v4i_t* as3_v4i;
__device__ __forceinline__ void async_b128(const void* g, void* l) {
    __builtin_amdgcn_global_load_async_to_lds_b128(
        (as1_v4i)(size_t)g, (as3_v4i)(size_t)l, 0, 0);
}
__device__ __forceinline__ void async_wait0() {
#if __has_builtin(__builtin_amdgcn_s_wait_asynccnt)
    __builtin_amdgcn_s_wait_asynccnt(0);
#else
    asm volatile("s_wait_asynccnt 0" ::: "memory");
#endif
}
__device__ __forceinline__ void async_wait4() {
#if __has_builtin(__builtin_amdgcn_s_wait_asynccnt)
    __builtin_amdgcn_s_wait_asynccnt(4);
#else
    asm volatile("s_wait_asynccnt 4" ::: "memory");
#endif
}
#endif

__device__ __forceinline__ int frag_kk(int v, int p, int kb) {
    return (v < 4) ? (kb + 2 * v + p) : (16 + kb + 2 * (v - 4) + p);
}

// fast activations: v_exp_f32 + v_rcp_f32, no IEEE divide expansion
__device__ __forceinline__ float fsigmoid(float x) {
    return __builtin_amdgcn_rcpf(1.f + __expf(-x));
}
__device__ __forceinline__ float ftanh(float x) {
    return 1.f - 2.f * __builtin_amdgcn_rcpf(1.f + __expf(2.f * x));
}

// ---------------------------------------------------------------------------
// Generic WMMA GEMM: C[M,N] = act(A[M,K](f16) * B[K,N](f16) + bias)
// Block tile 128x128, 8 waves (2x4 sub-tiles of 16x16 each), K staged 32-wide
// through double-buffered LDS with async DMA pipelined against WMMA.
// K must be a multiple of 32. ACT: 0=none, 1=sigmoid.
// ---------------------------------------------------------------------------
template <int ACT, int OUT16>
__global__ __launch_bounds__(256) void gemm_wmma(
    const h16* __restrict__ A, const h16* __restrict__ Bm,
    const float* __restrict__ bias,
    float* __restrict__ Cf, h16* __restrict__ Ch,
    int M, int Nstore, int Nphys, int K, int lda, int ldc)
{
    __shared__ __attribute__((aligned(16))) h16 As[2][128 * 32];
    __shared__ __attribute__((aligned(16))) h16 Bs[2][32 * 128];
    const int tid  = threadIdx.x;
    const int lane = tid & 31;
    const int wave = tid >> 5;
    const int wm = wave >> 1;      // 0..3  -> 32-row strip
    const int wn = wave & 1;       // 0..1  -> 64-col strip
    const int row0 = blockIdx.y * 128;
    const int col0 = blockIdx.x * 128;
    const int mlo = lane & 15;
    const int hig = lane >> 4;
    const int kb  = hig * 8;

    v8f acc[2][4];
#pragma unroll
    for (int i = 0; i < 2; ++i)
#pragma unroll
        for (int j = 0; j < 4; ++j)
#pragma unroll
            for (int e = 0; e < 8; ++e) acc[i][j][e] = 0.f;

    const int nK = K >> 5;

#if HAS_ASYNC_LDS
    // Each wave issues exactly 4 async-LDS ops per chunk (2 A + 2 B), so
    // s_wait_asynccnt 4 == "previous chunk landed, next still in flight".
    auto issue_chunk = [&](int kc, int buf) {
        const int k0 = kc << 5;
#pragma unroll
        for (int it = 0; it < 2; ++it) {        // A: 128 rows x 32 halves
            int idx = tid + it * 256;
            int r = idx >> 2, q = idx & 3;
            async_b128(A + (size_t)(row0 + r) * lda + k0 + q * 8,
                       &As[buf][r * 32 + q * 8]);
        }
#pragma unroll
        for (int it = 0; it < 2; ++it) {        // B: 32 rows x 128 halves
            int idx = tid + it * 256;
            int r = idx >> 4, q = idx & 15;
            async_b128(Bm + (size_t)(k0 + r) * Nphys + col0 + q * 8,
                       &Bs[buf][r * 128 + q * 8]);
        }
    };
    issue_chunk(0, 0);
#endif

    for (int kc = 0; kc < nK; ++kc) {
        const int buf = kc & 1;
#if HAS_ASYNC_LDS
        if (kc + 1 < nK) {
            issue_chunk(kc + 1, buf ^ 1);   // overlap next chunk's DMA
            async_wait4();                  // previous chunk complete
        } else {
            async_wait0();
        }
#else
        const int k0 = kc << 5;
        for (int idx = tid; idx < 128 * 32; idx += 256) {
            int r = idx >> 5, k = idx & 31;
            int gr = row0 + r, gk = k0 + k;
            As[buf][r * 32 + k] = (gr < M && gk < K) ? A[(size_t)gr * lda + gk] : (h16)0.f;
        }
        for (int idx = tid; idx < 32 * 128; idx += 256) {
            int r = idx >> 7, c = idx & 127;
            int gk = k0 + r, gc = col0 + c;
            Bs[buf][r * 128 + c] = (gk < K && gc < Nphys) ? Bm[(size_t)gk * Nphys + gc] : (h16)0.f;
        }
#endif
        __syncthreads();

        v16h af[2], bf[4];
#pragma unroll
        for (int i = 0; i < 2; ++i) {
            int rbase = wm * 32 + i * 16 + mlo;
#pragma unroll
            for (int v = 0; v < 8; ++v)
#pragma unroll
                for (int p = 0; p < 2; ++p)
                    af[i][2 * v + p] = As[buf][rbase * 32 + frag_kk(v, p, kb)];
        }
#pragma unroll
        for (int j = 0; j < 4; ++j) {
            int cbase = wn * 64 + j * 16 + mlo;
#pragma unroll
            for (int v = 0; v < 8; ++v)
#pragma unroll
                for (int p = 0; p < 2; ++p)
                    bf[j][2 * v + p] = Bs[buf][frag_kk(v, p, kb) * 128 + cbase];
        }
#pragma unroll
        for (int i = 0; i < 2; ++i)
#pragma unroll
            for (int j = 0; j < 4; ++j)
                acc[i][j] = __builtin_amdgcn_wmma_f32_16x16x32_f16(
                    false, af[i], false, bf[j], (short)0, acc[i][j], false, false);
        __syncthreads();   // all waves done reading buf before it is re-filled
    }

#pragma unroll
    for (int i = 0; i < 2; ++i)
#pragma unroll
        for (int j = 0; j < 4; ++j)
#pragma unroll
            for (int r = 0; r < 8; ++r) {
                int m = r + 8 * hig;
                int grow = row0 + wm * 32 + i * 16 + m;
                int gcol = col0 + wn * 64 + j * 16 + mlo;
                if (grow < M && gcol < Nstore) {
                    float v = acc[i][j][r];
                    if (bias) v += bias[gcol];
                    if (ACT == 1) v = fsigmoid(v);
                    if (OUT16) Ch[(size_t)grow * ldc + gcol] = (h16)v;
                    else       Cf[(size_t)grow * ldc + gcol] = v;
                }
            }
}

// ---------------------------------------------------------------------------
// GRU scan: single block, 8 waves. h (16x128) lives in LDS; Whh fragments
// (24 N-tiles of 16, 3 per wave, K=128 in 4 chunks) live in VGPRs for the
// whole scan. Per step: 12 v_wmma per wave -> gh in LDS -> fused gate math.
// xg already holds x*Wih^T + bih. Torch gate order r,z,n.
// ---------------------------------------------------------------------------
__global__ __launch_bounds__(256) void gru_scan(
    const float* __restrict__ xg,   // [16*NF_, 384], row = b*NF_+t
    const h16*  __restrict__ Whh,   // [128, 384] f16 row-major
    const float* __restrict__ bhh,  // [384]
    const float* __restrict__ h0,   // [16,128]
    h16*  __restrict__ ys,          // [16*NF_, 128] f16
    float* __restrict__ hlast)      // [16,128]
{
    __shared__ __attribute__((aligned(16))) h16   hb[16 * 128];
    __shared__ float hf[16 * 128];
    __shared__ float gh[16 * 384];
    const int tid  = threadIdx.x;
    const int lane = tid & 31;
    const int wave = tid >> 5;
    const int nl  = lane & 15;
    const int hig = lane >> 4;
    const int kb  = hig * 8;

    for (int i = tid; i < 2048; i += 256) {
        float v = h0[i];
        hf[i] = v; hb[i] = (h16)v;
    }

    // Whh B-fragments resident in VGPRs for the whole scan
    v16h bf[3][4];
#pragma unroll
    for (int j = 0; j < 3; ++j) {
        int n0 = (wave * 3 + j) * 16 + nl;
#pragma unroll
        for (int kc = 0; kc < 4; ++kc)
#pragma unroll
            for (int v = 0; v < 8; ++v)
#pragma unroll
                for (int p = 0; p < 2; ++p)
                    bf[j][kc][2 * v + p] = Whh[(size_t)(kc * 32 + frag_kk(v, p, kb)) * 384 + n0];
    }
    __syncthreads();

    for (int t = 0; t < NF_; ++t) {
        // A fragments (h as f16) from LDS: M=batch row = nl
        v16h af[4];
#pragma unroll
        for (int kc = 0; kc < 4; ++kc)
#pragma unroll
            for (int v = 0; v < 8; ++v)
#pragma unroll
                for (int p = 0; p < 2; ++p)
                    af[kc][2 * v + p] = hb[nl * 128 + kc * 32 + frag_kk(v, p, kb)];

#pragma unroll
        for (int j = 0; j < 3; ++j) {
            v8f acc;
#pragma unroll
            for (int e = 0; e < 8; ++e) acc[e] = 0.f;
#pragma unroll
            for (int kc = 0; kc < 4; ++kc)
                acc = __builtin_amdgcn_wmma_f32_16x16x32_f16(
                    false, af[kc], false, bf[j][kc], (short)0, acc, false, false);
            int n0 = (wave * 3 + j) * 16 + nl;
#pragma unroll
            for (int r = 0; r < 8; ++r)
                gh[(r + 8 * hig) * 384 + n0] = acc[r];
        }
        __syncthreads();

        for (int i = tid; i < 2048; i += 256) {
            int b = i >> 7, jj = i & 127;
            const float* xr = xg + ((size_t)b * NF_ + t) * 384;
            if (t + 1 < NF_)  // overlap next step's gate-input fetch with WMMA
                __builtin_prefetch(xg + ((size_t)b * NF_ + t + 1) * 384 + jj, 0, 1);
            float ghr = gh[b * 384 + jj]        + bhh[jj];
            float ghz = gh[b * 384 + 128 + jj]  + bhh[128 + jj];
            float ghn = gh[b * 384 + 256 + jj]  + bhh[256 + jj];
            float r = fsigmoid(xr[jj] + ghr);
            float z = fsigmoid(xr[128 + jj] + ghz);
            float n = ftanh(xr[256 + jj] + r * ghn);
            float hn = (1.f - z) * n + z * hf[i];
            hf[i] = hn; hb[i] = (h16)hn;
            ys[((size_t)b * NF_ + t) * 128 + jj] = (h16)hn;
        }
        __syncthreads();
    }
    for (int i = tid; i < 2048; i += 256) hlast[i] = hf[i];
}

// ---------------------------------------------------------------------------
// Elementwise / prep kernels
// ---------------------------------------------------------------------------
__global__ void k_frames(const float* __restrict__ x, h16* __restrict__ fr) {
    size_t idx = (size_t)blockIdx.x * 256 + threadIdx.x;
    if (idx >= (size_t)BF_ * FRAME_) return;
    int row = (int)(idx >> 10), n = (int)(idx & 1023);
    int b = row / NF_, t = row % NF_;
    float w = 0.5f * (1.f - __cosf(2.f * PI_ * (float)n / (float)FRAME_));
    fr[idx] = (h16)(x[(size_t)b * TLEN_ + (size_t)t * HOP_ + n] * w);
}

__global__ void k_dft_basis(h16* __restrict__ bd) {
    size_t idx = (size_t)blockIdx.x * 256 + threadIdx.x;
    if (idx >= (size_t)FRAME_ * SPECW_) return;
    int n = (int)(idx / SPECW_), j = (int)(idx % SPECW_);
    float v = 0.f;
    if (j < FREQ_)                 v =  __cosf(2.f * PI_ * (float)n * (float)j / (float)FRAME_);
    else if (j >= FP_ && j < FP_ + FREQ_) {
        int k = j - FP_;           v = -__sinf(2.f * PI_ * (float)n * (float)k / (float)FRAME_);
    }
    bd[idx] = (h16)v;
}

__global__ void k_idft_basis(float* __restrict__ bi) {
    size_t idx = (size_t)blockIdx.x * 256 + threadIdx.x;
    if (idx >= (size_t)SPECW_ * FRAME_) return;
    int r = (int)(idx / FRAME_), n = (int)(idx % FRAME_);
    float v = 0.f;
    if (r < FREQ_) {
        float w = (r == 0 || r == 512) ? 1.f : 2.f;
        v = w * __cosf(2.f * PI_ * (float)r * (float)n / (float)FRAME_) / (float)FRAME_;
    } else if (r >= FP_ && r < FP_ + FREQ_) {
        int k = r - FP_;
        float w = (k == 0 || k == 512) ? 1.f : 2.f;
        v = -w * __sinf(2.f * PI_ * (float)k * (float)n / (float)FRAME_) / (float)FRAME_;
    }
    bi[idx] = v;
}

// Bcomb[r][e] = sum_n iDFT[r][n] * enc_W[e][n]   (fuse irfft with 1x1 encoder)
__global__ void k_comb(const float* __restrict__ bi, const float* __restrict__ encW,
                       h16* __restrict__ bc) {
    size_t idx = (size_t)blockIdx.x * 256 + threadIdx.x;
    if (idx >= (size_t)SPECW_ * ENC_) return;
    int r = (int)(idx / ENC_), e = (int)(idx % ENC_);
    float s = 0.f;
    for (int n = 0; n < FRAME_; ++n)
        s += bi[(size_t)r * FRAME_ + n] * encW[(size_t)e * FRAME_ + n];
    bc[idx] = (h16)s;
}

// dst[k][n] = src[n][k], f32 -> f16, zero-padded to Kpad x Npad
__global__ void k_transpose_pad(const float* __restrict__ src, h16* __restrict__ dst,
                                int K, int Kpad, int N, int Npad) {
    size_t idx = (size_t)blockIdx.x * 256 + threadIdx.x;
    if (idx >= (size_t)Kpad * Npad) return;
    int k = (int)(idx / Npad), n = (int)(idx % Npad);
    dst[idx] = (k < K && n < N) ? (h16)src[(size_t)n * K + k] : (h16)0.f;
}

__global__ void k_mag(const float* __restrict__ spec, h16* __restrict__ mag) {
    size_t idx = (size_t)blockIdx.x * 256 + threadIdx.x;
    if (idx >= (size_t)BF_ * FPK_) return;
    int row = (int)(idx / FPK_), j = (int)(idx % FPK_);
    float v = 0.f;
    if (j < FREQ_) {
        float re = spec[(size_t)row * SPECW_ + j];
        float im = spec[(size_t)row * SPECW_ + FP_ + j];
        v = sqrtf(fmaxf(re * re + im * im, 1.1920929e-7f));
    }
    mag[idx] = (h16)v;
}

__global__ void k_apply_mask1(const float* __restrict__ mask, const float* __restrict__ spec,
                              h16* __restrict__ est) {
    size_t idx = (size_t)blockIdx.x * 256 + threadIdx.x;
    if (idx >= (size_t)BF_ * FP_) return;
    int row = (int)(idx / FP_), j = (int)(idx % FP_);
    float re = 0.f, im = 0.f;
    if (j < FREQ_) {
        float m = mask[(size_t)row * FP_ + j];
        re = m * spec[(size_t)row * SPECW_ + j];
        im = m * spec[(size_t)row * SPECW_ + FP_ + j];
    }
    est[(size_t)row * SPECW_ + j]       = (h16)re;
    est[(size_t)row * SPECW_ + FP_ + j] = (h16)im;
}

__global__ __launch_bounds__(256) void k_iln(const float* __restrict__ enc,
                                             const float* __restrict__ gamma,
                                             const float* __restrict__ beta,
                                             h16* __restrict__ out) {
    __shared__ float red[256];
    int row = blockIdx.x, tid = threadIdx.x;
    float v = enc[(size_t)row * ENC_ + tid];
    red[tid] = v; __syncthreads();
    for (int s = 128; s > 0; s >>= 1) { if (tid < s) red[tid] += red[tid + s]; __syncthreads(); }
    float mean = red[0] / (float)ENC_; __syncthreads();
    float d = v - mean;
    red[tid] = d * d; __syncthreads();
    for (int s = 128; s > 0; s >>= 1) { if (tid < s) red[tid] += red[tid + s]; __syncthreads(); }
    float var = red[0] / (float)ENC_;
    out[(size_t)row * ENC_ + tid] = (h16)(d * rsqrtf(var + 1e-7f) * gamma[tid] + beta[tid]);
}

__global__ void k_est2(const float* __restrict__ mask, const float* __restrict__ enc,
                       h16* __restrict__ est2) {
    size_t idx = (size_t)blockIdx.x * 256 + threadIdx.x;
    if (idx >= (size_t)BF_ * ENC_) return;
    est2[idx] = (h16)(mask[idx] * enc[idx]);
}

// Overlap-add as a gather (<=4 frames per sample): deterministic, no atomics.
__global__ void k_ola(const float* __restrict__ dec, float* __restrict__ out) {
    size_t idx = (size_t)blockIdx.x * 256 + threadIdx.x;
    if (idx >= (size_t)BDIM_ * TLEN_) return;
    int b = (int)(idx >> 18);
    int s = (int)(idx & (TLEN_ - 1));
    int tmax = s >> 8; if (tmax > NF_ - 1) tmax = NF_ - 1;
    int tmin = (s < FRAME_) ? 0 : ((s - (FRAME_ - 1) + (HOP_ - 1)) >> 8);
    float acc = 0.f;
    for (int t = tmin; t <= tmax; ++t)
        acc += dec[((size_t)b * NF_ + t) * FRAME_ + (s - (t << 8))];
    out[idx] = acc;
}

// ---------------------------------------------------------------------------
// Host launcher
// ---------------------------------------------------------------------------
static inline int cdiv_(long long a, long long b) { return (int)((a + b - 1) / b); }

extern "C" void kernel_launch(void* const* d_in, const int* in_sizes, int n_in,
                              void* d_out, int out_size, void* d_ws, size_t ws_size,
                              hipStream_t stream) {
    const float* x        = (const float*)d_in[0];
    const float* in_st1   = (const float*)d_in[1];
    const float* in_st2   = (const float*)d_in[2];
    const float* s1_Wih1  = (const float*)d_in[3];
    const float* s1_Whh1  = (const float*)d_in[4];
    const float* s1_bih1  = (const float*)d_in[5];
    const float* s1_bhh1  = (const float*)d_in[6];
    const float* s1_Wih2  = (const float*)d_in[7];
    const float* s1_Whh2  = (const float*)d_in[8];
    const float* s1_bih2  = (const float*)d_in[9];
    const float* s1_bhh2  = (const float*)d_in[10];
    const float* s1_Wd    = (const float*)d_in[11];
    const float* s1_bd    = (const float*)d_in[12];
    const float* enc_W    = (const float*)d_in[13];
    const float* gamma    = (const float*)d_in[14];
    const float* beta     = (const float*)d_in[15];
    const float* s2_Wih1  = (const float*)d_in[16];
    const float* s2_Whh1  = (const float*)d_in[17];
    const float* s2_bih1  = (const float*)d_in[18];
    const float* s2_bhh1  = (const float*)d_in[19];
    const float* s2_Wih2  = (const float*)d_in[20];
    const float* s2_Whh2  = (const float*)d_in[21];
    const float* s2_bih2  = (const float*)d_in[22];
    const float* s2_bhh2  = (const float*)d_in[23];
    const float* s2_Wd    = (const float*)d_in[24];
    const float* s2_bd    = (const float*)d_in[25];
    const float* dec_W    = (const float*)d_in[26];
    float* out = (float*)d_out;
    const size_t OUT_SAMPLES = (size_t)BDIM_ * TLEN_;

    // ---- carve workspace ----
    size_t off = 0;
    auto carve = [&](size_t bytes) -> void* {
        void* p = (char*)d_ws + off;
        off += (bytes + 255) & ~(size_t)255;
        return p;
    };
    h16*   frames16 = (h16*)  carve((size_t)BF_ * FRAME_ * 2);
    h16*   Bdft16   = (h16*)  carve((size_t)FRAME_ * SPECW_ * 2);
    float* specf    = (float*)carve((size_t)BF_ * SPECW_ * 4);   // later reused as dec output
    h16*   mag16    = (h16*)  carve((size_t)BF_ * FPK_ * 2);     // later reused as est2
    float* xg       = (float*)carve((size_t)BF_ * 384 * 4);
    h16*   ysA      = (h16*)  carve((size_t)BF_ * HID_ * 2);
    h16*   ysB      = (h16*)  carve((size_t)BF_ * HID_ * 2);
    float* mask1f   = (float*)carve((size_t)BF_ * FP_ * 4);      // later reused as mask2
    h16*   est16    = (h16*)  carve((size_t)BF_ * SPECW_ * 2);
    float* Bidft    = (float*)carve((size_t)SPECW_ * FRAME_ * 4);
    h16*   Bcomb16  = (h16*)  carve((size_t)SPECW_ * ENC_ * 2);
    float* encf     = (float*)carve((size_t)BF_ * ENC_ * 4);
    h16*   encn16   = (h16*)  carve((size_t)BF_ * ENC_ * 2);
    h16*   W1s1     = (h16*)  carve((size_t)FPK_ * 384 * 2);
    h16*   Whh1s1   = (h16*)  carve((size_t)HID_ * 384 * 2);
    h16*   W2s1     = (h16*)  carve((size_t)HID_ * 384 * 2);
    h16*   Whh2s1   = (h16*)  carve((size_t)HID_ * 384 * 2);
    h16*   Wd1      = (h16*)  carve((size_t)HID_ * FP_ * 2);
    h16*   W1s2     = (h16*)  carve((size_t)ENC_ * 384 * 2);
    h16*   Whh1s2   = (h16*)  carve((size_t)HID_ * 384 * 2);
    h16*   W2s2     = (h16*)  carve((size_t)HID_ * 384 * 2);
    h16*   Whh2s2   = (h16*)  carve((size_t)HID_ * 384 * 2);
    h16*   Wd2      = (h16*)  carve((size_t)HID_ * ENC_ * 2);
    h16*   Bdec     = (h16*)  carve((size_t)ENC_ * FRAME_ * 2);
    float* decf   = specf;     // alias: spec dead after mask1 apply
    float* mask2f = mask1f;    // alias: mask1 dead after apply
    h16*   est2_16 = mag16;    // alias: mag dead after xg1 GEMM

    // ---- prep: bases + f16 weight transposes ----
    k_dft_basis <<<cdiv_((long long)FRAME_ * SPECW_, 256), 256, 0, stream>>>(Bdft16);
    k_idft_basis<<<cdiv_((long long)SPECW_ * FRAME_, 256), 256, 0, stream>>>(Bidft);
    k_comb      <<<cdiv_((long long)SPECW_ * ENC_, 256), 256, 0, stream>>>(Bidft, enc_W, Bcomb16);
    k_transpose_pad<<<cdiv_((long long)FPK_ * 384, 256), 256, 0, stream>>>(s1_Wih1, W1s1, FREQ_, FPK_, 384, 384);
    k_transpose_pad<<<cdiv_((long long)HID_ * 384, 256), 256, 0, stream>>>(s1_Whh1, Whh1s1, HID_, HID_, 384, 384);
    k_transpose_pad<<<cdiv_((long long)HID_ * 384, 256), 256, 0, stream>>>(s1_Wih2, W2s1,   HID_, HID_, 384, 384);
    k_transpose_pad<<<cdiv_((long long)HID_ * 384, 256), 256, 0, stream>>>(s1_Whh2, Whh2s1, HID_, HID_, 384, 384);
    k_transpose_pad<<<cdiv_((long long)HID_ * FP_, 256), 256, 0, stream>>>(s1_Wd, Wd1, HID_, HID_, FREQ_, FP_);
    k_transpose_pad<<<cdiv_((long long)ENC_ * 384, 256), 256, 0, stream>>>(s2_Wih1, W1s2, ENC_, ENC_, 384, 384);
    k_transpose_pad<<<cdiv_((long long)HID_ * 384, 256), 256, 0, stream>>>(s2_Whh1, Whh1s2, HID_, HID_, 384, 384);
    k_transpose_pad<<<cdiv_((long long)HID_ * 384, 256), 256, 0, stream>>>(s2_Wih2, W2s2,   HID_, HID_, 384, 384);
    k_transpose_pad<<<cdiv_((long long)HID_ * 384, 256), 256, 0, stream>>>(s2_Whh2, Whh2s2, HID_, HID_, 384, 384);
    k_transpose_pad<<<cdiv_((long long)HID_ * ENC_, 256), 256, 0, stream>>>(s2_Wd, Wd2, HID_, HID_, ENC_, ENC_);
    k_transpose_pad<<<cdiv_((long long)ENC_ * FRAME_, 256), 256, 0, stream>>>(dec_W, Bdec, ENC_, ENC_, FRAME_, FRAME_);

    // ---- stage 1: framing + windowing, rFFT as WMMA GEMM ----
    k_frames<<<cdiv_((long long)BF_ * FRAME_, 256), 256, 0, stream>>>(x, frames16);
    {
        dim3 g(cdiv_(SPECW_, 128), cdiv_(BF_, 128));
        gemm_wmma<0, 0><<<g, 256, 0, stream>>>(frames16, Bdft16, nullptr, specf, nullptr,
                                               BF_, SPECW_, SPECW_, FRAME_, FRAME_, SPECW_);
    }
    k_mag<<<cdiv_((long long)BF_ * FPK_, 256), 256, 0, stream>>>(specf, mag16);

    // ---- separation block 1 ----
    {
        dim3 g(cdiv_(384, 128), cdiv_(BF_, 128));
        gemm_wmma<0, 0><<<g, 256, 0, stream>>>(mag16, W1s1, s1_bih1, xg, nullptr,
                                               BF_, 384, 384, FPK_, FPK_, 384);
    }
    gru_scan<<<1, 256, 0, stream>>>(xg, Whh1s1, s1_bhh1, in_st1, ysA,
                                    out + OUT_SAMPLES);
    {
        dim3 g(cdiv_(384, 128), cdiv_(BF_, 128));
        gemm_wmma<0, 0><<<g, 256, 0, stream>>>(ysA, W2s1, s1_bih2, xg, nullptr,
                                               BF_, 384, 384, HID_, HID_, 384);
    }
    gru_scan<<<1, 256, 0, stream>>>(xg, Whh2s1, s1_bhh2, in_st1 + 2048, ysB,
                                    out + OUT_SAMPLES + 2048);
    {
        dim3 g(cdiv_(FREQ_, 128), cdiv_(BF_, 128));
        gemm_wmma<1, 0><<<g, 256, 0, stream>>>(ysB, Wd1, s1_bd, mask1f, nullptr,
                                               BF_, FREQ_, FP_, HID_, HID_, FP_);
    }
    k_apply_mask1<<<cdiv_((long long)BF_ * FP_, 256), 256, 0, stream>>>(mask1f, specf, est16);

    // ---- fused irFFT + encoder, instant layer norm ----
    {
        dim3 g(cdiv_(ENC_, 128), cdiv_(BF_, 128));
        gemm_wmma<0, 0><<<g, 256, 0, stream>>>(est16, Bcomb16, nullptr, encf, nullptr,
                                               BF_, ENC_, ENC_, SPECW_, SPECW_, ENC_);
    }
    k_iln<<<BF_, 256, 0, stream>>>(encf, gamma, beta, encn16);

    // ---- separation block 2 ----
    {
        dim3 g(cdiv_(384, 128), cdiv_(BF_, 128));
        gemm_wmma<0, 0><<<g, 256, 0, stream>>>(encn16, W1s2, s2_bih1, xg, nullptr,
                                               BF_, 384, 384, ENC_, ENC_, 384);
    }
    gru_scan<<<1, 256, 0, stream>>>(xg, Whh1s2, s2_bhh1, in_st2, ysA,
                                    out + OUT_SAMPLES + 4096);
    {
        dim3 g(cdiv_(384, 128), cdiv_(BF_, 128));
        gemm_wmma<0, 0><<<g, 256, 0, stream>>>(ysA, W2s2, s2_bih2, xg, nullptr,
                                               BF_, 384, 384, HID_, HID_, 384);
    }
    gru_scan<<<1, 256, 0, stream>>>(xg, Whh2s2, s2_bhh2, in_st2 + 2048, ysB,
                                    out + OUT_SAMPLES + 6144);
    {
        dim3 g(cdiv_(ENC_, 128), cdiv_(BF_, 128));
        gemm_wmma<1, 0><<<g, 256, 0, stream>>>(ysB, Wd2, s2_bd, mask2f, nullptr,
                                               BF_, ENC_, ENC_, HID_, HID_, ENC_);
    }
    k_est2<<<cdiv_((long long)BF_ * ENC_, 256), 256, 0, stream>>>(mask2f, encf, est2_16);

    // ---- decoder GEMM + overlap-add ----
    {
        dim3 g(cdiv_(FRAME_, 128), cdiv_(BF_, 128));
        gemm_wmma<0, 0><<<g, 256, 0, stream>>>(est2_16, Bdec, nullptr, decf, nullptr,
                                               BF_, FRAME_, FRAME_, ENC_, ENC_, FRAME_);
    }
    k_ola<<<cdiv_((long long)BDIM_ * TLEN_, 256), 256, 0, stream>>>(decf, out);
}
